// Pyramid_64510408786181
// MI455X (gfx1250) — matrix operand order repaired and checked by reading, not compiled, not run on hardware
//
#include <hip/hip_runtime.h>

// ---------------------------------------------------------------------------
// Types for CDNA5 WMMA (wave32): v16bf A/B fragments, v8f C/D accumulator.
// ---------------------------------------------------------------------------
typedef __attribute__((ext_vector_type(16))) __bf16 v16bf;
typedef __attribute__((ext_vector_type(8)))  __bf16 v8bf;
typedef __attribute__((ext_vector_type(8)))  float  v8f;

#define V8F_ZERO {0.f,0.f,0.f,0.f,0.f,0.f,0.f,0.f}

__device__ __forceinline__ v8f wmma_bf16(v16bf a, v16bf b, v8f c) {
  // v_wmma_f32_16x16x32_bf16  D = A(16x32) * B(32x16) + C
  return __builtin_amdgcn_wmma_f32_16x16x32_bf16(false, a, false, b, (short)0, c,
                                                 false, false);
}

// A-matrix 16-bit layout (ISA 7.12.2): lane l -> row M=l%16, kk=(l<16)?0:8.
// elements 0..7  = K(kk..kk+7)      -> 16B at byte offset 2*(32*s+kk)
// elements 8..15 = K(16+kk..16+kk+7)-> 16B at byte offset 2*(32*s+16+kk)
__device__ __forceinline__ v16bf load_a_frag(const __bf16* rowptr, int s, int kk) {
  v8bf lo = *(const v8bf*)(rowptr + s * 32 + kk);
  v8bf hi = *(const v8bf*)(rowptr + s * 32 + 16 + kk);
  return __builtin_shufflevector(lo, hi, 0,1,2,3,4,5,6,7,8,9,10,11,12,13,14,15);
}

__device__ __forceinline__ float sigf(float x) { return 1.0f / (1.0f + __expf(-x)); }
__device__ __forceinline__ float tanh_fast(float x) {
  x = fminf(fmaxf(x, -15.f), 15.f);
  float t = __expf(2.f * x);
  return (t - 1.f) / (t + 1.f);
}

// ---------------------------------------------------------------------------
// Pack an f32 (K x N, row-major) weight into WMMA-B-fragment order, bf16.
// B-matrix 16-bit layout: lane l -> col N=l%16, K base=(l<16)?0:8, elem i -> K=base+i (i<8), 16+base+(i-8) (i>=8).
// Fragment (tile t, k-step s, lane l) = 16 contiguous bf16 (32 bytes):
//   dst[((t*Ksteps + s)*32 + l)*16 + i] = src[(s*32 + (i<8 ? base+i : 16+base+i-8))*N + (t*16 + l%16)]
// ---------------------------------------------------------------------------
__global__ void pack_wmma_b(const float* __restrict__ src, __bf16* __restrict__ dst,
                            int N, int Ksteps) {
  int idx = blockIdx.x * blockDim.x + threadIdx.x;
  int total = (N / 16) * Ksteps * 32 * 16;
  if (idx >= total) return;
  int i    = idx & 15;
  int l    = (idx >> 4) & 31;
  int rest = idx >> 9;           // t*Ksteps + s
  int s    = rest % Ksteps;
  int t    = rest / Ksteps;
  int base = (l < 16) ? 0 : 8;
  int n    = t * 16 + (l & 15);
  int k    = s * 32 + ((i < 8) ? (base + i) : (16 + base + i - 8));
  dst[idx] = (__bf16)src[(size_t)k * N + n];
}

// ---------------------------------------------------------------------------
// Pyramid kernel: one block per batch element, 256 threads (8 waves).
// Dynamic LDS layout (bytes) — Abuf FIRST so per-step A-fragment offsets fold
// into the 16-bit DS immediate (row base < 32KB):
//   Abuf    0      : 64*256 bf16 (WMMA A tile, rows = pairs, K = concat(hl,hr))
//   sA      32768  : 64*256 f32  (states ping)
//   sB      98304  : 64*256 f32  (states pong)
//   comp    163840 : 64*256 f32  (TreeLSTM output [h|c])
//   selw    229376 : 256 f32
//   logit   230400 : 64 f32
//   pbuf    230656 : 64 f32
//   wlb     230912 : 64 f32  (copy_left coeff)
//   wrb     231168 : 64 f32  (copy_right coeff)
//   part    231424 : 256 f32 (logit partial sums)
// total 232448
// ---------------------------------------------------------------------------
#define PYR_LDS_BYTES 232448

__global__ __launch_bounds__(256, 1)
void pyramid_kernel(const int* __restrict__ sentences,
                    const float* __restrict__ emb,
                    const __bf16* __restrict__ packW,   // 40 tiles x 8 ksteps
                    const float* __restrict__ comp_b,
                    const float* __restrict__ sel_w,
                    const float* __restrict__ sel_b,
                    float* __restrict__ hfinal) {
  extern __shared__ __align__(16) char smem[];
  __bf16* Abuf  = (__bf16*)(smem);
  float*  sA    = (float*)(smem + 32768);
  float*  sB    = (float*)(smem + 98304);
  float*  comp  = (float*)(smem + 163840);
  float*  selw  = (float*)(smem + 229376);
  float*  logit = (float*)(smem + 230400);
  float*  pbuf  = (float*)(smem + 230656);
  float*  wlb   = (float*)(smem + 230912);
  float*  wrb   = (float*)(smem + 231168);
  float*  part  = (float*)(smem + 231424);

  const int tid = threadIdx.x;
  const int b   = blockIdx.x;
  const int ln  = tid & 31;
  const int wv  = tid >> 5;           // 0..7 -> gate block [wv*16, wv*16+16)
  const int col = ln & 15;
  const int kk  = (ln < 16) ? 0 : 8;  // A-fragment K interleave
  const int mh  = (ln < 16) ? 0 : 8;  // C/D row offset

  // ---- init: states = emb[tokens], preload sel_w ----
  {
    int j  = tid >> 2;
    int k0 = (tid & 3) * 64;
    int tok = sentences[b * 64 + j];
    const float* erow = emb + (size_t)tok * 256;
    for (int k = k0; k < k0 + 64; ++k) sA[j * 256 + k] = erow[k];
    selw[tid] = sel_w[tid];
  }
  __syncthreads();

  float* cur = sA;
  float* nxt = sB;

  const int n_g = wv * 16 + col;      // gate index 0..127 owned by this lane
  const float bI  = comp_b[n_g];
  const float bFl = comp_b[128 + n_g];
  const float bFr = comp_b[256 + n_g];
  const float bG  = comp_b[384 + n_g];
  const float bO  = comp_b[512 + n_g];
  const float selb = sel_b[0];

  for (int n = 63; n >= 1; --n) {     // n = number of adjacent pairs this layer
    // ---- build A: row j = [h_left(128) | h_right(128)] as bf16, zero-pad ----
    {
      int j  = tid >> 2;
      int k0 = (tid & 3) * 64;
      for (int k = k0; k < k0 + 64; ++k) {
        float v = 0.f;
        if (j < n) v = (k < 128) ? cur[j * 256 + k] : cur[(j + 1) * 256 + (k - 128)];
        Abuf[j * 256 + k] = (__bf16)v;
      }
    }
    __syncthreads();

    // ---- GEMM (64 x 640 x 256) fused with TreeLSTM gates ----
    // Wave wv owns gate columns [wv*16, wv*16+16): N-tiles {wv, wv+8, wv+16, wv+24, wv+32}
    // => i/fl/fr/g/o for the SAME (row m, gate n_g) land in the SAME lane+slot.
    for (int mt = 0; mt < 4; ++mt) {
      v8f aI = V8F_ZERO, aFl = V8F_ZERO, aFr = V8F_ZERO, aG = V8F_ZERO, aO = V8F_ZERO;
      const __bf16* arow = Abuf + (mt * 16 + col) * 256;
      // Software pipeline: A fragment for step s+1 is in flight during the
      // 5-WMMA group of step s (hides ds_load latency behind XDL work).
      v16bf a = load_a_frag(arow, 0, kk);
#pragma unroll
      for (int s = 0; s < 8; ++s) {
        v16bf an = a;
        if (s < 7) an = load_a_frag(arow, s + 1, kk);
        v16bf b0 = *(const v16bf*)(packW + (size_t)(((wv +  0) * 8 + s) * 32 + ln) * 16);
        v16bf b1 = *(const v16bf*)(packW + (size_t)(((wv +  8) * 8 + s) * 32 + ln) * 16);
        v16bf b2 = *(const v16bf*)(packW + (size_t)(((wv + 16) * 8 + s) * 32 + ln) * 16);
        v16bf b3 = *(const v16bf*)(packW + (size_t)(((wv + 24) * 8 + s) * 32 + ln) * 16);
        v16bf b4 = *(const v16bf*)(packW + (size_t)(((wv + 32) * 8 + s) * 32 + ln) * 16);
        aI  = wmma_bf16(a, b0, aI);
        aFl = wmma_bf16(a, b1, aFl);
        aFr = wmma_bf16(a, b2, aFr);
        aG  = wmma_bf16(a, b3, aG);
        aO  = wmma_bf16(a, b4, aO);
        a = an;
      }
#pragma unroll
      for (int r = 0; r < 8; ++r) {
        int m  = mt * 16 + mh + r;
        int mr = (m + 1 < 64) ? (m + 1) : 63;     // padding rows: value unused
        float clv = cur[m  * 256 + 128 + n_g];
        float crv = cur[mr * 256 + 128 + n_g];
        float iv  = aI[r]  + bI;
        float flv = aFl[r] + bFl;
        float frv = aFr[r] + bFr;
        float gv  = aG[r]  + bG;
        float ov  = aO[r]  + bO;
        float cnew = sigf(flv) * clv + sigf(frv) * crv + sigf(iv) * tanh_fast(gv);
        float hnew = sigf(ov) * tanh_fast(cnew);
        comp[m * 256 + n_g]       = hnew;
        comp[m * 256 + 128 + n_g] = cnew;
      }
    }
    __syncthreads();

    // ---- selection logits: 4 threads per pair, LDS partial reduce ----
    {
      int j  = tid >> 2;
      int k0 = (tid & 3) * 64;
      float s = 0.f;
      for (int k = k0; k < k0 + 64; ++k) s += comp[j * 256 + k] * selw[k];
      part[tid] = s;
    }
    __syncthreads();
    if (tid < n)
      logit[tid] = selb + part[4*tid] + part[4*tid+1] + part[4*tid+2] + part[4*tid+3];
    __syncthreads();

    // ---- softmax + exclusive prefix/suffix sums (n <= 63: serial is cheap) ----
    if (tid == 0) {
      float mx = logit[0];
      for (int j = 1; j < n; ++j) mx = fmaxf(mx, logit[j]);
      float sum = 0.f;
      for (int j = 0; j < n; ++j) { float e = __expf(logit[j] - mx); pbuf[j] = e; sum += e; }
      float inv = 1.f / sum;
      float cs = 0.f;
      for (int j = 0; j < n; ++j) { float p = pbuf[j] * inv; pbuf[j] = p; wrb[j] = cs; cs += p; }
      float rcs = 0.f;
      for (int j = n - 1; j >= 0; --j) { wlb[j] = rcs; rcs += pbuf[j]; }
    }
    __syncthreads();

    // ---- state update into the other buffer ----
    {
      int j  = tid >> 2;
      int k0 = (tid & 3) * 64;
      if (j < n) {
        float wL = wlb[j], wR = wrb[j], pP = pbuf[j];
        for (int k = k0; k < k0 + 64; ++k)
          nxt[j * 256 + k] = wL * cur[j * 256 + k] + wR * cur[(j + 1) * 256 + k]
                           + pP * comp[j * 256 + k];
      }
    }
    __syncthreads();
    float* t = cur; cur = nxt; nxt = t;
  }

  hfinal[b * 256 + tid] = cur[tid];
}

// ---------------------------------------------------------------------------
// MLP kernel: 1 block, 256 threads. 64x1024x256 and 64x1024x1024 bf16 WMMA
// GEMMs with fused bias+ReLU into bf16 LDS activations, then 1024->3 VALU GEMM.
// LDS: hA 32768 | z0 131072 | z1 131072  => 294912 bytes (A buffers first =>
// DS-immediate-foldable fragment offsets).
// ---------------------------------------------------------------------------
#define MLP_LDS_BYTES 294912

__global__ __launch_bounds__(256, 1)
void mlp_kernel(const float* __restrict__ hfinal,
                const __bf16* __restrict__ packW0,  // 64 tiles x 8 ksteps
                const __bf16* __restrict__ packW1,  // 64 tiles x 32 ksteps
                const float* __restrict__ b0,
                const float* __restrict__ b1,
                const float* __restrict__ cls_w,
                const float* __restrict__ cls_b,
                float* __restrict__ out) {
  extern __shared__ __align__(16) char smem[];
  __bf16* hA = (__bf16*)(smem);
  __bf16* z0 = (__bf16*)(smem + 32768);
  __bf16* z1 = (__bf16*)(smem + 32768 + 131072);

  const int tid = threadIdx.x;
  const int ln  = tid & 31;
  const int wv  = tid >> 5;
  const int col = ln & 15;
  const int kk  = (ln < 16) ? 0 : 8;
  const int mh  = (ln < 16) ? 0 : 8;

  for (int e = tid; e < 64 * 256; e += 256) hA[e] = (__bf16)hfinal[e];
  __syncthreads();

  // ---- GEMM1: z0 = relu(h @ w0 + b0), 64x1024, K=256 (8 k-steps) ----
  for (int nt = wv; nt < 64; nt += 8) {
    int n = nt * 16 + col;
    float bias = b0[n];
    for (int mt = 0; mt < 4; ++mt) {
      v8f acc = V8F_ZERO;
      const __bf16* arow = hA + (mt * 16 + col) * 256;
      v16bf a = load_a_frag(arow, 0, kk);
#pragma unroll
      for (int s = 0; s < 8; ++s) {
        v16bf an = a;
        if (s < 7) an = load_a_frag(arow, s + 1, kk);
        v16bf w = *(const v16bf*)(packW0 + (size_t)((nt * 8 + s) * 32 + ln) * 16);
        acc = wmma_bf16(a, w, acc);
        a = an;
      }
#pragma unroll
      for (int r = 0; r < 8; ++r) {
        float z = fmaxf(acc[r] + bias, 0.f);
        z0[(mt * 16 + mh + r) * 1024 + n] = (__bf16)z;
      }
    }
  }
  __syncthreads();

  // ---- GEMM2: z1 = relu(z0 @ w1 + b1), 64x1024, K=1024 (32 k-steps) ----
  for (int nt = wv; nt < 64; nt += 8) {
    int n = nt * 16 + col;
    float bias = b1[n];
    for (int mt = 0; mt < 4; ++mt) {
      v8f acc = V8F_ZERO;
      const __bf16* arow = z0 + (mt * 16 + col) * 1024;
      v16bf a = load_a_frag(arow, 0, kk);
#pragma unroll 8
      for (int s = 0; s < 32; ++s) {
        v16bf an = a;
        if (s < 31) an = load_a_frag(arow, s + 1, kk);
        v16bf w = *(const v16bf*)(packW1 + (size_t)((nt * 32 + s) * 32 + ln) * 16);
        acc = wmma_bf16(a, w, acc);
        a = an;
      }
#pragma unroll
      for (int r = 0; r < 8; ++r) {
        float z = fmaxf(acc[r] + bias, 0.f);
        z1[(mt * 16 + mh + r) * 1024 + n] = (__bf16)z;
      }
    }
  }
  __syncthreads();

  // ---- classifier: out = z1 @ cls_w + cls_b (64x3, K=1024) ----
  if (tid < 192) {
    int bo = tid / 3, c = tid % 3;
    float s = cls_b[c];
    for (int k = 0; k < 1024; ++k)
      s += (float)z1[bo * 1024 + k] * cls_w[k * 3 + c];
    out[bo * 3 + c] = s;
  }
}

// ---------------------------------------------------------------------------
// Workspace layout (bytes):
//   packW  @ 0        : 640*256  bf16 = 327680
//   packW0 @ 327680   : 1024*256 bf16 = 524288
//   packW1 @ 851968   : 1024*1024 bf16 = 2097152
//   hfinal @ 2949120  : 64*256 f32    = 65536
// ---------------------------------------------------------------------------
extern "C" void kernel_launch(void* const* d_in, const int* in_sizes, int n_in,
                              void* d_out, int out_size, void* d_ws, size_t ws_size,
                              hipStream_t stream) {
  (void)in_sizes; (void)n_in; (void)out_size; (void)ws_size;
  const int*   sentences = (const int*)d_in[0];
  // d_in[1] transitions: unused by the reference computation
  const float* emb    = (const float*)d_in[2];
  const float* comp_W = (const float*)d_in[3];
  const float* comp_b = (const float*)d_in[4];
  const float* sel_w  = (const float*)d_in[5];
  const float* sel_b  = (const float*)d_in[6];
  const float* w0     = (const float*)d_in[7];
  const float* b0     = (const float*)d_in[8];
  const float* w1     = (const float*)d_in[9];
  const float* b1     = (const float*)d_in[10];
  const float* clsw   = (const float*)d_in[11];
  const float* clsb   = (const float*)d_in[12];

  char* ws = (char*)d_ws;
  __bf16* packW  = (__bf16*)(ws);
  __bf16* packW0 = (__bf16*)(ws + 327680);
  __bf16* packW1 = (__bf16*)(ws + 851968);
  float*  hfin   = (float*)(ws + 2949120);

  // Allow >64KB dynamic LDS (320KB/WGP on CDNA5). Host-side attribute set,
  // not a stream op — safe under graph capture; errors ignored.
  (void)hipFuncSetAttribute((const void*)pyramid_kernel,
                            hipFuncAttributeMaxDynamicSharedMemorySize, PYR_LDS_BYTES);
  (void)hipFuncSetAttribute((const void*)mlp_kernel,
                            hipFuncAttributeMaxDynamicSharedMemorySize, MLP_LDS_BYTES);

  // Re-pack weights every call (d_ws is not preserved between calls).
  pack_wmma_b<<<640,  256, 0, stream>>>(comp_W, packW,  640,  8);   // 40t x 8s
  pack_wmma_b<<<1024, 256, 0, stream>>>(w0,     packW0, 1024, 8);   // 64t x 8s
  pack_wmma_b<<<4096, 256, 0, stream>>>(w1,     packW1, 1024, 32);  // 64t x 32s

  pyramid_kernel<<<64, 256, PYR_LDS_BYTES, stream>>>(sentences, emb, packW,
                                                     comp_b, sel_w, sel_b, hfin);
  mlp_kernel<<<1, 256, MLP_LDS_BYTES, stream>>>(hfin, packW0, packW1,
                                                b0, b1, clsw, clsb, (float*)d_out);
}
